// TritonFlashAttention_39779987096072
// MI455X (gfx1250) — matrix operand order repaired and checked by reading, not compiled, not run on hardware
//
#include <hip/hip_runtime.h>

// ---------------- problem constants ----------------
constexpr int B_  = 2;
constexpr int S_  = 2048;
constexpr int D_  = 1024;
constexpr int H_  = 16;
constexpr int HD_ = 64;   // head dim
constexpr int N3_ = 3 * D_;

typedef __attribute__((ext_vector_type(16))) _Float16 v16h;
typedef __attribute__((ext_vector_type(8)))  float    v8f;
typedef __attribute__((ext_vector_type(2)))  __fp16   v2fp;   // cvt_pkrtz result type
typedef __attribute__((ext_vector_type(4)))  int      v4i;

union Frag { uint4 u[2]; v16h v; };
union PkCvt { v2fp h[2]; uint2 u; };

// Optional CDNA5 async global->LDS path (ASYNCcnt), guarded for toolchain.
#if defined(__has_builtin)
#if __has_builtin(__builtin_amdgcn_global_load_async_to_lds_b128) && \
    __has_builtin(__builtin_amdgcn_s_wait_asynccnt)
#define USE_ASYNC_LDS 1
#endif
#endif

#ifdef USE_ASYNC_LDS
typedef __attribute__((address_space(1))) v4i gv4i;   // global int4
typedef __attribute__((address_space(3))) v4i lv4i;   // LDS int4
#endif

// =====================================================================
// Kernel 1: QKV GEMM.  qkv = x @ Wqkv + bqkv  (f32 in, f16 out, f32 acc)
// Writes Q (pre-scaled by 1/sqrt(HD)), K, V into workspace as [B,H,S,HD].
// Block: 256 thr (8 waves). Tile 128x128, K-step 32.
// =====================================================================
__global__ __launch_bounds__(256) void qkv_gemm_kernel(
    const float* __restrict__ x,      // [B*S, D]
    const float* __restrict__ Wqkv,   // [D, 3D]
    const float* __restrict__ bqkv,   // [3D]
    _Float16* __restrict__ qws,       // [B,H,S,HD]
    _Float16* __restrict__ kws,
    _Float16* __restrict__ vws)
{
  __shared__ __align__(16) _Float16 As[128 * 32];   // [m][k]
  __shared__ __align__(16) _Float16 Bs[128 * 32];   // transposed: [n][k]

  const int m0   = blockIdx.x * 128;
  const int n0   = blockIdx.y * 128;
  const int tid  = threadIdx.x;
  const int lane = tid & 31;
  const int wave = tid >> 5;
  const int half = lane >> 4;
  const int l16  = lane & 15;

  const int rowBase = (wave & 3) * 32;   // 0,32,64,96
  const int colBase = (wave >> 2) * 64;  // 0,64

  v8f acc[2][4] = {};

  for (int k0 = 0; k0 < D_; k0 += 32) {
    __syncthreads();
    // A tile: 128x32 f32 as 1024 float4 loads, packed-convert, b64 LDS stores
    for (int i = 0; i < 4; ++i) {
      int idx = tid + 256 * i;                 // 0..1023
      int r = idx >> 3, c4 = (idx & 7) * 4;
      float4 f = *(const float4*)&x[(size_t)(m0 + r) * D_ + k0 + c4];
      PkCvt cv;
      cv.h[0] = __builtin_amdgcn_cvt_pkrtz(f.x, f.y);
      cv.h[1] = __builtin_amdgcn_cvt_pkrtz(f.z, f.w);
      *(uint2*)&As[r * 32 + c4] = cv.u;
    }
    // B tile: Wqkv[k][n] float4 along n (coalesced) -> Bs[n][k]
    for (int i = 0; i < 4; ++i) {
      int idx = tid + 256 * i;                 // 0..1023
      int kk = idx >> 5, nf = (idx & 31) * 4;
      float4 f = *(const float4*)&Wqkv[(size_t)(k0 + kk) * N3_ + n0 + nf];
      Bs[(nf + 0) * 32 + kk] = (_Float16)f.x;
      Bs[(nf + 1) * 32 + kk] = (_Float16)f.y;
      Bs[(nf + 2) * 32 + kk] = (_Float16)f.z;
      Bs[(nf + 3) * 32 + kk] = (_Float16)f.w;
    }
    __syncthreads();

    Frag a[2], b[4];
    for (int rt = 0; rt < 2; ++rt) {
      int m = rowBase + rt * 16 + l16;
      a[rt].u[0] = *(const uint4*)&As[m * 32 + half * 8];
      a[rt].u[1] = *(const uint4*)&As[m * 32 + 16 + half * 8];
    }
    for (int ct = 0; ct < 4; ++ct) {
      int n = colBase + ct * 16 + l16;
      b[ct].u[0] = *(const uint4*)&Bs[n * 32 + half * 8];
      b[ct].u[1] = *(const uint4*)&Bs[n * 32 + 16 + half * 8];
    }
    for (int rt = 0; rt < 2; ++rt)
      for (int ct = 0; ct < 4; ++ct)
        acc[rt][ct] = __builtin_amdgcn_wmma_f32_16x16x32_f16(
            false, a[rt].v, false, b[ct].v, (short)0, acc[rt][ct], false, false);
  }

  // Epilogue: bias, scale-Q, scatter to [B,H,S,HD] f16
  for (int rt = 0; rt < 2; ++rt) {
    for (int ct = 0; ct < 4; ++ct) {
      int c     = n0 + colBase + ct * 16 + l16;   // column in [0,3D)
      int which = c / D_;                          // 0=q,1=k,2=v (tile-uniform)
      int rem   = c - which * D_;
      int h     = rem / HD_;
      int hd    = rem - h * HD_;
      float bias = bqkv[c];
      _Float16* dst = (which == 0) ? qws : (which == 1) ? kws : vws;
      float sc = (which == 0) ? 0.125f : 1.0f;     // fold sm_scale into Q
      for (int r = 0; r < 8; ++r) {
        int grow = m0 + rowBase + rt * 16 + r + 8 * half;
        int bb = grow / S_;
        int s  = grow - bb * S_;
        float vout = (acc[rt][ct][r] + bias) * sc;
        dst[((size_t)(bb * H_ + h) * S_ + s) * HD_ + hd] = (_Float16)vout;
      }
    }
  }
}

// =====================================================================
// Kernel 2: Flash attention. Grid = (S/128, B*H), 256 thr (8 waves).
// Q A-fragments preloaded to registers (loop-invariant, direct global).
// K tile via async global->LDS when available; online softmax in f32;
// P staged through per-wave LDS to become a WMMA A-frag.
// =====================================================================
__global__ __launch_bounds__(256) void flash_attn_kernel(
    const _Float16* __restrict__ qws,   // [B,H,S,HD] pre-scaled
    const _Float16* __restrict__ kws,
    const _Float16* __restrict__ vws,
    const float*    __restrict__ mask,  // [H,S,S]
    _Float16* __restrict__ ctxws)       // [B,S,D]
{
  __shared__ __align__(16) _Float16 Ks[64 * 64];      // [key][hd]  (== Bt for QK^T)
  __shared__ __align__(16) _Float16 Vt[64 * 64];      // [hd][key]  (== Bt for PV)
  __shared__ __align__(16) _Float16 Ps[8][16 * 64];   // per-wave P tile [m][key]

  const int bh   = blockIdx.y;
  const int b    = bh / H_;
  const int h    = bh - b * H_;
  const int q0   = blockIdx.x * 128;
  const int tid  = threadIdx.x;
  const int lane = tid & 31;
  const int wave = tid >> 5;
  const int half = lane >> 4;
  const int l16  = lane & 15;
  const int qBase = wave * 16;

  // Preload Q A-fragments for this wave's 16 rows (loop-invariant).
  Frag qf[2];
  {
    const _Float16* qg = qws + ((size_t)bh * S_ + q0 + qBase + l16) * HD_;
    for (int kk = 0; kk < 2; ++kk) {
      qf[kk].u[0] = *(const uint4*)(qg + kk * 32 + half * 8);
      qf[kk].u[1] = *(const uint4*)(qg + kk * 32 + 16 + half * 8);
    }
  }

  v8f Oacc[4] = {};
  float mrow[8], lrow[8];
  for (int r = 0; r < 8; ++r) { mrow[r] = -1e30f; lrow[r] = 0.0f; }

  const int NKB = S_ / 64;
  for (int kb = 0; kb < NKB; ++kb) {
    const int k0 = kb * 64;
    __syncthreads();
    // K tile: contiguous 8KB copy, layout [key][hd]
    {
      const _Float16* src = kws + ((size_t)bh * S_ + k0) * HD_;
#ifdef USE_ASYNC_LDS
      for (int i = 0; i < 2; ++i) {
        int idx = tid + 256 * i;                  // 0..511 x 16B
        __builtin_amdgcn_global_load_async_to_lds_b128(
            (gv4i*)(src + idx * 8), (lv4i*)&Ks[idx * 8], 0, 0);
      }
#else
      const uint4* s4 = (const uint4*)src;
      uint4* d4 = (uint4*)Ks;
      for (int i = 0; i < 2; ++i) d4[tid + 256 * i] = s4[tid + 256 * i];
#endif
    }
    // V tile: transpose into Vt[hd][key] (b128 loads, b16 scattered stores)
    {
      const uint4* src = (const uint4*)(vws + ((size_t)bh * S_ + k0) * HD_);
      for (int i = 0; i < 2; ++i) {
        int idx = tid + 256 * i;                  // 64 keys x 8 chunks
        int key = idx >> 3, c8 = (idx & 7) * 8;
        union { uint4 u; _Float16 f[8]; } cv; cv.u = src[idx];
        for (int j = 0; j < 8; ++j) Vt[(c8 + j) * 64 + key] = cv.f[j];
      }
    }
    // Prefetch next K/V block into L2 (global_prefetch_b8)
    if (kb + 1 < NKB) {
      const char* nk = (const char*)(kws + ((size_t)bh * S_ + k0 + 64) * HD_);
      const char* nv = (const char*)(vws + ((size_t)bh * S_ + k0 + 64) * HD_);
      __builtin_prefetch(nk + tid * 32, 0, 1);
      __builtin_prefetch(nv + tid * 32, 0, 1);
    }
#ifdef USE_ASYNC_LDS
    __builtin_amdgcn_s_wait_asynccnt(0);
#endif
    __syncthreads();

    // ---- S = Q K^T (16 rows x 64 keys per wave) ----
    v8f Sacc[4] = {};
    for (int kk = 0; kk < 2; ++kk) {
      for (int ct = 0; ct < 4; ++ct) {
        Frag bk;
        int n = ct * 16 + l16;
        bk.u[0] = *(const uint4*)&Ks[n * 64 + kk * 32 + half * 8];
        bk.u[1] = *(const uint4*)&Ks[n * 64 + kk * 32 + 16 + half * 8];
        Sacc[ct] = __builtin_amdgcn_wmma_f32_16x16x32_f16(
            false, qf[kk].v, false, bk.v, (short)0, Sacc[ct], false, false);
      }
    }

    // ---- add mask ----
    const float* mbase = mask + ((size_t)h * S_ + (q0 + qBase)) * S_ + k0;
    for (int ct = 0; ct < 4; ++ct)
      for (int r = 0; r < 8; ++r) {
        int qrow = r + 8 * half;
        Sacc[ct][r] += mbase[(size_t)qrow * S_ + ct * 16 + l16];
      }

    // ---- online softmax (row = r + 8*half, reduce across 16-lane group) ----
    float mnew[8], alpha[8], rsum[8];
    for (int r = 0; r < 8; ++r) {
      float v = fmaxf(fmaxf(Sacc[0][r], Sacc[1][r]), fmaxf(Sacc[2][r], Sacc[3][r]));
      for (int off = 1; off < 16; off <<= 1) v = fmaxf(v, __shfl_xor(v, off, 32));
      mnew[r]  = fmaxf(mrow[r], v);
      alpha[r] = __expf(mrow[r] - mnew[r]);
      rsum[r]  = 0.0f;
    }
    for (int ct = 0; ct < 4; ++ct)
      for (int r = 0; r < 8; ++r) {
        float p = __expf(Sacc[ct][r] - mnew[r]);
        rsum[r] += p;
        Ps[wave][(r + 8 * half) * 64 + ct * 16 + l16] = (_Float16)p;
      }
    for (int r = 0; r < 8; ++r) {
      float v = rsum[r];
      for (int off = 1; off < 16; off <<= 1) v += __shfl_xor(v, off, 32);
      lrow[r] = lrow[r] * alpha[r] + v;
      mrow[r] = mnew[r];
    }
    for (int ot = 0; ot < 4; ++ot)
      for (int r = 0; r < 8; ++r) Oacc[ot][r] *= alpha[r];

    // ---- O += P V ----
    for (int kk = 0; kk < 2; ++kk) {
      Frag ap;
      ap.u[0] = *(const uint4*)&Ps[wave][l16 * 64 + kk * 32 + half * 8];
      ap.u[1] = *(const uint4*)&Ps[wave][l16 * 64 + kk * 32 + 16 + half * 8];
      for (int ot = 0; ot < 4; ++ot) {
        Frag bv;
        int n = ot * 16 + l16;
        bv.u[0] = *(const uint4*)&Vt[n * 64 + kk * 32 + half * 8];
        bv.u[1] = *(const uint4*)&Vt[n * 64 + kk * 32 + 16 + half * 8];
        Oacc[ot] = __builtin_amdgcn_wmma_f32_16x16x32_f16(
            false, ap.v, false, bv.v, (short)0, Oacc[ot], false, false);
      }
    }
  }

  // ---- normalize and store ctx as f16 [B,S,D] ----
  for (int ot = 0; ot < 4; ++ot)
    for (int r = 0; r < 8; ++r) {
      int s = q0 + qBase + r + 8 * half;
      float o = Oacc[ot][r] / lrow[r];
      ctxws[(size_t)(b * S_ + s) * D_ + h * HD_ + ot * 16 + l16] = (_Float16)o;
    }
}

// =====================================================================
// Kernel 3: Output GEMM.  out = ctx @ Wout + bout   (f16 A, f32 out)
// A-fragments loaded directly from global f16 (16B aligned runs).
// =====================================================================
__global__ __launch_bounds__(256) void out_gemm_kernel(
    const _Float16* __restrict__ ctx,  // [B*S, D] f16
    const float* __restrict__ Wout,    // [D, D]
    const float* __restrict__ bout,    // [D]
    float* __restrict__ out)           // [B*S, D]
{
  __shared__ __align__(16) _Float16 Bs[128 * 32];   // [n][k]

  const int m0   = blockIdx.x * 128;
  const int n0   = blockIdx.y * 128;
  const int tid  = threadIdx.x;
  const int lane = tid & 31;
  const int wave = tid >> 5;
  const int half = lane >> 4;
  const int l16  = lane & 15;
  const int rowBase = (wave & 3) * 32;
  const int colBase = (wave >> 2) * 64;

  v8f acc[2][4] = {};

  for (int k0 = 0; k0 < D_; k0 += 32) {
    __syncthreads();
    // B tile: Wout[k][n] float4 along n -> Bs[n][k]
    for (int i = 0; i < 4; ++i) {
      int idx = tid + 256 * i;
      int kk = idx >> 5, nf = (idx & 31) * 4;
      float4 f = *(const float4*)&Wout[(size_t)(k0 + kk) * D_ + n0 + nf];
      Bs[(nf + 0) * 32 + kk] = (_Float16)f.x;
      Bs[(nf + 1) * 32 + kk] = (_Float16)f.y;
      Bs[(nf + 2) * 32 + kk] = (_Float16)f.z;
      Bs[(nf + 3) * 32 + kk] = (_Float16)f.w;
    }
    __syncthreads();

    Frag a[2], b[4];
    for (int rt = 0; rt < 2; ++rt) {
      const _Float16* ag = ctx + (size_t)(m0 + rowBase + rt * 16 + l16) * D_ + k0;
      a[rt].u[0] = *(const uint4*)(ag + half * 8);
      a[rt].u[1] = *(const uint4*)(ag + 16 + half * 8);
    }
    for (int ct = 0; ct < 4; ++ct) {
      int n = colBase + ct * 16 + l16;
      b[ct].u[0] = *(const uint4*)&Bs[n * 32 + half * 8];
      b[ct].u[1] = *(const uint4*)&Bs[n * 32 + 16 + half * 8];
    }
    for (int rt = 0; rt < 2; ++rt)
      for (int ct = 0; ct < 4; ++ct)
        acc[rt][ct] = __builtin_amdgcn_wmma_f32_16x16x32_f16(
            false, a[rt].v, false, b[ct].v, (short)0, acc[rt][ct], false, false);
  }

  for (int rt = 0; rt < 2; ++rt)
    for (int ct = 0; ct < 4; ++ct) {
      int col = n0 + colBase + ct * 16 + l16;
      float bias = bout[col];
      for (int r = 0; r < 8; ++r) {
        int grow = m0 + rowBase + rt * 16 + r + 8 * half;
        out[(size_t)grow * D_ + col] = acc[rt][ct][r] + bias;
      }
    }
}

// =====================================================================
// Host launcher
// =====================================================================
extern "C" void kernel_launch(void* const* d_in, const int* in_sizes, int n_in,
                              void* d_out, int out_size, void* d_ws, size_t ws_size,
                              hipStream_t stream) {
  const float* x     = (const float*)d_in[0];   // [B,S,D]
  const float* amask = (const float*)d_in[1];   // [1,H,S,S]
  const float* Wqkv  = (const float*)d_in[2];   // [D,3D]
  const float* bqkv  = (const float*)d_in[3];   // [3D]
  const float* Wout  = (const float*)d_in[4];   // [D,D]
  const float* bout  = (const float*)d_in[5];   // [D]
  float* out = (float*)d_out;

  const size_t NQ = (size_t)B_ * S_ * D_;       // 4,194,304 per buffer
  _Float16* qws   = (_Float16*)d_ws;
  _Float16* kws   = qws + NQ;
  _Float16* vws   = kws + NQ;
  _Float16* ctxws = vws + NQ;

  // 1) QKV projection
  qkv_gemm_kernel<<<dim3((B_ * S_) / 128, N3_ / 128), 256, 0, stream>>>(
      x, Wqkv, bqkv, qws, kws, vws);

  // 2) Flash attention
  flash_attn_kernel<<<dim3(S_ / 128, B_ * H_), 256, 0, stream>>>(
      qws, kws, vws, amask, ctxws);

  // 3) Output projection
  out_gemm_kernel<<<dim3((B_ * S_) / 128, D_ / 128), 256, 0, stream>>>(
      ctxws, Wout, bout, out);
}
